// MambaBlock_85383949845390
// MI455X (gfx1250) — matrix operand order repaired and checked by reading, not compiled, not run on hardware
//
#include <hip/hip_runtime.h>
#include <hip/hip_bf16.h>

typedef __attribute__((ext_vector_type(16))) _Float16 v16h;
typedef __attribute__((ext_vector_type(8)))  float    v8f;

// ---------------------------------------------------------------------------
// WMMA helpers (CDNA5 / gfx1250, wave32): D = A(16x32 f16) * B(32x16 f16) + C
// ---------------------------------------------------------------------------
__device__ __forceinline__ v8f wmma_f32_16x16x32(v16h a, v16h b, v8f c) {
  return __builtin_amdgcn_wmma_f32_16x16x32_f16(
      /*neg_a=*/false, a, /*neg_b=*/false, b,
      /*c_mod=*/(short)0, c, /*reuse_a=*/false, /*reuse_b=*/false);
}

// A fragment: 16 rows (M) x 32 cols (K) from row-major [M][stride] f16 array.
// ISA layout: lane m=lane&15; kbase=(lane>>4)*8; elems 0..7 -> k=kbase+e,
// elems 8..15 -> k=16+kbase+(e-8).
__device__ __forceinline__ v16h load_frag_a(const _Float16* src, int stride, int lane) {
  int m  = lane & 15;
  int kb = (lane >> 4) * 8;
  const _Float16* r = src + m * stride + kb;
  v16h a;
#pragma unroll
  for (int e = 0; e < 8; ++e) a[e] = r[e];
#pragma unroll
  for (int e = 0; e < 8; ++e) a[8 + e] = r[16 + e];
  return a;
}

// B fragment: logical B is 32(K) x 16(N); src laid out [N][stride] so that
// element (k,n) = src[n*stride + k] (B^T row-major -> contiguous in K).
// ISA layout: lane n=lane&15; kbase=(lane>>4)*8; elems 0..7 -> k=kbase+e,
// elems 8..15 -> k=16+kbase+(e-8).
__device__ __forceinline__ v16h load_frag_b(const _Float16* src_nk, int stride, int lane) {
  int n  = lane & 15;
  int kb = (lane >> 4) * 8;
  const _Float16* r = src_nk + n * stride + kb;
  v16h b;
#pragma unroll
  for (int e = 0; e < 8; ++e) b[e] = r[e];
#pragma unroll
  for (int e = 0; e < 8; ++e) b[8 + e] = r[16 + e];
  return b;
}

__device__ __forceinline__ float silu_f(float x) { return x / (1.f + __expf(-x)); }

// ---------------------------------------------------------------------------
// f16 GEMM: C[M,N] = A[M,K] * B[N,K]^T (+ optional f32 residual)
// 256 threads = 8 waves; macro tile 128x128; per-wave 32x64 (2x4 WMMA tiles).
// Double-buffered LDS ping-pong: global b128 loads for tile k+1 issued before
// the WMMA block of tile k; L2 prefetch for tile k+2.
// Branchless staging: OOB rows are clamped (their products are discarded by
// the guarded store), so the hot loop has no exec-mask branching.
// ---------------------------------------------------------------------------
__global__ void __launch_bounds__(256)
gemm_f16_kernel(const _Float16* __restrict__ A, const _Float16* __restrict__ Bm,
                float* __restrict__ C, const float* __restrict__ resid,
                int M, int N, int K) {
  __shared__ _Float16 As[2][128][40];  // 80B row stride: 16B aligned, 20-dword pitch
  __shared__ _Float16 Bs[2][128][40];
  int tid = threadIdx.x, lane = tid & 31, wid = tid >> 5;
  int m0 = blockIdx.x * 128, n0 = blockIdx.y * 128;
  int wm = wid >> 1;      // 0..3 : 32-row band
  int wn = wid & 1;       // 0..1 : 64-col band
  v8f acc[2][4] = {};

  // staging: each thread owns half of one row (16 halves = 2 x b128)
  int srow = tid >> 1;
  int scol = (tid & 1) * 16;
  int gmA = m0 + srow; if (gmA > M - 1) gmA = M - 1;   // clamp, no zero-fill
  int gnB = n0 + srow; if (gnB > N - 1) gnB = N - 1;
  const _Float16* aSrc = A  + (size_t)gmA * K + scol;
  const _Float16* bSrc = Bm + (size_t)gnB * K + scol;

  int T = K >> 5;  // K / 32
  // prologue: stage tile 0 into buffer 0
  {
    uint4 a0v = *(const uint4*)(aSrc);
    uint4 a1v = *(const uint4*)(aSrc + 8);
    uint4 b0v = *(const uint4*)(bSrc);
    uint4 b1v = *(const uint4*)(bSrc + 8);
    *(uint4*)&As[0][srow][scol]     = a0v;
    *(uint4*)&As[0][srow][scol + 8] = a1v;
    *(uint4*)&Bs[0][srow][scol]     = b0v;
    *(uint4*)&Bs[0][srow][scol + 8] = b1v;
  }

  int buf = 0;
  for (int kt = 0; kt < T; ++kt) {
    __syncthreads();
    uint4 a0v = {}, a1v = {}, b0v = {}, b1v = {};
    bool pre = (kt + 1 < T);
    if (pre) {
      const _Float16* ap = aSrc + (size_t)(kt + 1) * 32;
      const _Float16* bp = bSrc + (size_t)(kt + 1) * 32;
      a0v = *(const uint4*)ap; a1v = *(const uint4*)(ap + 8);
      b0v = *(const uint4*)bp; b1v = *(const uint4*)(bp + 8);
      if (kt + 2 < T) {
        __builtin_prefetch(aSrc + (size_t)(kt + 2) * 32, 0, 1);  // global_prefetch_b8
        __builtin_prefetch(bSrc + (size_t)(kt + 2) * 32, 0, 1);
      }
    }
    // compute tile kt from LDS buffer `buf`
    v16h fa0 = load_frag_a(&As[buf][wm * 32 +  0][0], 40, lane);
    v16h fa1 = load_frag_a(&As[buf][wm * 32 + 16][0], 40, lane);
#pragma unroll
    for (int j = 0; j < 4; ++j) {
      v16h fb = load_frag_b(&Bs[buf][wn * 64 + j * 16][0], 40, lane);
      acc[0][j] = wmma_f32_16x16x32(fa0, fb, acc[0][j]);
      acc[1][j] = wmma_f32_16x16x32(fa1, fb, acc[1][j]);
    }
    if (pre) {
      *(uint4*)&As[buf ^ 1][srow][scol]     = a0v;
      *(uint4*)&As[buf ^ 1][srow][scol + 8] = a1v;
      *(uint4*)&Bs[buf ^ 1][srow][scol]     = b0v;
      *(uint4*)&Bs[buf ^ 1][srow][scol + 8] = b1v;
    }
    buf ^= 1;
  }

  int nl = lane & 15, mb = (lane >> 4) * 8;
#pragma unroll
  for (int i = 0; i < 2; ++i)
#pragma unroll
    for (int j = 0; j < 4; ++j) {
      int gn = n0 + wn * 64 + j * 16 + nl;
      if (gn >= N) continue;
#pragma unroll
      for (int r = 0; r < 8; ++r) {
        int gm = m0 + wm * 32 + i * 16 + mb + r;
        if (gm >= M) continue;
        size_t off = (size_t)gm * N + gn;
        float v = acc[i][j][r];
        if (resid) v += resid[off];
        C[off] = v;
      }
    }
}

// ---------------------------------------------------------------------------
// Elementwise / prep kernels
// ---------------------------------------------------------------------------
__global__ void f32_to_f16_kernel(const float* __restrict__ src,
                                  _Float16* __restrict__ dst, size_t n) {
  size_t i = (size_t)blockIdx.x * blockDim.x + threadIdx.x;
  if (i < n) dst[i] = (_Float16)src[i];
}

// depthwise causal conv (K=4) over xBC columns of zxbcdt, then silu
__global__ void conv_silu_kernel(const float* __restrict__ zx,
                                 const float* __restrict__ conv_w,
                                 float* __restrict__ xconv) {
  size_t idx = (size_t)blockIdx.x * blockDim.x + threadIdx.x;
  if (idx >= (size_t)4096 * 2304) return;
  int ch = (int)(idx % 2304);
  size_t row = idx / 2304;        // b*L + l
  int l = (int)(row & 2047);
  float acc = 0.f;
#pragma unroll
  for (int j = 0; j < 4; ++j) {
    int li = l + j - 3;
    if (li >= 0) {
      size_t r2 = row + (size_t)(j - 3);
      acc += zx[r2 * 4384 + 2048 + ch] * conv_w[ch * 4 + j];
    }
  }
  xconv[idx] = silu_f(acc);
}

// X_dt = silu(x_branch)*dt  (f16, chunk layout (B,H,c,s,P)); also dtbuf
__global__ void xdt_kernel(const float* __restrict__ zx,
                           const float* __restrict__ xconv,
                           const float* __restrict__ dt_bias,
                           _Float16* __restrict__ Xh, float* __restrict__ dtbuf) {
  size_t idx = (size_t)blockIdx.x * blockDim.x + threadIdx.x;
  if (idx >= (size_t)4096 * 2048) return;
  int p = (int)(idx & 63);
  int h = (int)((idx >> 6) & 31);
  size_t row = idx >> 11;         // b*L + l
  int l = (int)(row & 2047), b = (int)(row >> 11);
  float draw = zx[row * 4384 + 4352 + h] + dt_bias[h];
  float dt = (draw > 20.f) ? draw : log1pf(__expf(draw));
  float X = silu_f(xconv[row * 2304 + h * 64 + p]);
  int ci = l >> 6, s = l & 63;
  Xh[(((size_t)(b * 32 + h)) * 32 + ci) * 4096 + s * 64 + p] = (_Float16)(X * dt);
  if (p == 0) dtbuf[row * 32 + h] = dt;
}

// silu(B), silu(C) -> f16 chunk layout (B,c,s,N)
__global__ void bc_silu_kernel(const float* __restrict__ xconv,
                               _Float16* __restrict__ Bh, _Float16* __restrict__ Ch) {
  size_t idx = (size_t)blockIdx.x * blockDim.x + threadIdx.x;
  if (idx >= (size_t)4096 * 128) return;
  int n = (int)(idx & 127);
  size_t row = idx >> 7;
  int l = (int)(row & 2047), b = (int)(row >> 11);
  int ci = l >> 6, s = l & 63;
  size_t o = ((size_t)(b * 32 + ci)) * 8192 + s * 128 + n;
  Bh[o] = (_Float16)silu_f(xconv[row * 2304 + 2048 + n]);
  Ch[o] = (_Float16)silu_f(xconv[row * 2304 + 2176 + n]);
}

// per (b,h,chunk) inclusive cumsum of A*dt over the 64 positions
__global__ void __launch_bounds__(64)
cumsum_kernel(const float* __restrict__ dtbuf, const float* __restrict__ A_log,
              float* __restrict__ acum) {
  int bid = blockIdx.x;           // ((b*32+h)*32+ci)
  int ci = bid & 31, h = (bid >> 5) & 31, b = bid >> 10;
  __shared__ float sh[64];
  int s = threadIdx.x;
  int l = ci * 64 + s;
  sh[s] = -__expf(A_log[h]) * dtbuf[((size_t)b * 2048 + l) * 32 + h];
  __syncthreads();
  float acc = 0.f;
  for (int i = 0; i <= s; ++i) acc += sh[i];
  acum[(((size_t)(b * 32 + h)) * 32 + ci) * 64 + s] = acc;
}

// ---------------------------------------------------------------------------
// Per-(b,chunk,head) intra-chunk SSD: G=C*B^T, masked decay, Y_diag, states
// 128 threads = 4 waves; each wave owns one 16-row band.
// ---------------------------------------------------------------------------
__global__ void __launch_bounds__(128)
ssd_chunk_kernel(const _Float16* __restrict__ Ch, const _Float16* __restrict__ Bh,
                 const _Float16* __restrict__ Xh, const float* __restrict__ acum,
                 float* __restrict__ Ybuf, float* __restrict__ states) {
  int bid = blockIdx.x;
  int h = bid & 31, ci = (bid >> 5) & 31, b = bid >> 10;
  __shared__ _Float16 Cld[64][136];   // [l][n] (272B pitch: 16B aligned, 68 dwords)
  __shared__ _Float16 Bld[64][136];   // [s][n]
  __shared__ _Float16 BT[128][66];    // [n][s]
  __shared__ _Float16 XT[64][66];     // [p][s]
  __shared__ _Float16 XTs[64][66];    // [p][s] * decay(s)
  __shared__ _Float16 Ml[64][66];     // masked decay matrix [l][s]
  __shared__ float Ac[64];
  __shared__ float dec[64];
  int tid = threadIdx.x, lane = tid & 31, wid = tid >> 5;

  const float* ac = acum + (((size_t)(b * 32 + h)) * 32 + ci) * 64;
  if (tid < 64) Ac[tid] = ac[tid];
  __syncthreads();
  if (tid < 64) dec[tid] = __expf(Ac[63] - Ac[tid]);

  const uint4* Cg4 = (const uint4*)(Ch + ((size_t)(b * 32 + ci)) * 8192);
  const uint4* Bg4 = (const uint4*)(Bh + ((size_t)(b * 32 + ci)) * 8192);
  for (int idx = tid; idx < 1024; idx += 128) {  // 8192 halves = 1024 uint4
    int s = idx >> 3, n8 = (idx & 7) * 8;
    uint4 cv = Cg4[idx];
    uint4 bv = Bg4[idx];
    *(uint4*)&Cld[s][n8] = cv;
    *(uint4*)&Bld[s][n8] = bv;
    const _Float16* hv = (const _Float16*)&bv;
#pragma unroll
    for (int e = 0; e < 8; ++e) BT[n8 + e][s] = hv[e];
  }
  __syncthreads();  // dec + C/B ready
  const uint4* Xg4 = (const uint4*)(Xh + (((size_t)(b * 32 + h)) * 32 + ci) * 4096);
  for (int idx = tid; idx < 512; idx += 128) {   // 4096 halves = 512 uint4
    int s = idx >> 3, p8 = (idx & 7) * 8;
    uint4 xv = Xg4[idx];
    const _Float16* hv = (const _Float16*)&xv;
    float ds = dec[s];
#pragma unroll
    for (int e = 0; e < 8; ++e) {
      XT[p8 + e][s]  = hv[e];
      XTs[p8 + e][s] = (_Float16)((float)hv[e] * ds);
    }
  }
  __syncthreads();

  int L0 = wid * 16;
  // ---- G[l,s] = sum_n C[l,n]*B[s,n], then mask * exp(Ac[l]-Ac[s]) -> Ml
  {
    v8f g[4] = {};
    for (int n0 = 0; n0 < 128; n0 += 32) {
      v16h a = load_frag_a(&Cld[L0][n0], 136, lane);
#pragma unroll
      for (int t = 0; t < 4; ++t) {
        v16h bf = load_frag_b(&Bld[t * 16][n0], 136, lane);
        g[t] = wmma_f32_16x16x32(a, bf, g[t]);
      }
    }
    int sc = lane & 15, mb = (lane >> 4) * 8;
#pragma unroll
    for (int t = 0; t < 4; ++t) {
      int s = t * 16 + sc;
#pragma unroll
      for (int r = 0; r < 8; ++r) {
        int l = L0 + mb + r;
        float v = (s <= l) ? g[t][r] * __expf(Ac[l] - Ac[s]) : 0.f;
        Ml[l][s] = (_Float16)v;
      }
    }
  }
  __syncthreads();

  // ---- Y_diag[l,p] = sum_s Ml[l,s]*X[s,p]
  {
    v8f yd[4] = {};
    for (int s0 = 0; s0 < 64; s0 += 32) {
      v16h a = load_frag_a(&Ml[L0][s0], 66, lane);
#pragma unroll
      for (int t = 0; t < 4; ++t) {
        v16h bf = load_frag_b(&XT[t * 16][s0], 66, lane);
        yd[t] = wmma_f32_16x16x32(a, bf, yd[t]);
      }
    }
    int pc = lane & 15, mb = (lane >> 4) * 8;
#pragma unroll
    for (int t = 0; t < 4; ++t) {
      int p = t * 16 + pc;
#pragma unroll
      for (int r = 0; r < 8; ++r) {
        int l = L0 + mb + r;
        Ybuf[((size_t)b * 2048 + ci * 64 + l) * 2048 + h * 64 + p] = yd[t][r];
      }
    }
  }

  // ---- states[p,n] = sum_s dec(s)*X[s,p]*B[s,n]
  {
    v8f st[8] = {};
    int P0 = wid * 16;
    for (int s0 = 0; s0 < 64; s0 += 32) {
      v16h a = load_frag_a(&XTs[P0][s0], 66, lane);
#pragma unroll
      for (int t = 0; t < 8; ++t) {
        v16h bf = load_frag_b(&BT[t * 16][s0], 66, lane);
        st[t] = wmma_f32_16x16x32(a, bf, st[t]);
      }
    }
    int nc = lane & 15, mb = (lane >> 4) * 8;
    float* sg = states + (((size_t)(b * 32 + h)) * 32 + ci) * 8192;
#pragma unroll
    for (int t = 0; t < 8; ++t) {
      int n = t * 16 + nc;
#pragma unroll
      for (int r = 0; r < 8; ++r) sg[(P0 + mb + r) * 128 + n] = st[t][r];
    }
  }
}

// ---------------------------------------------------------------------------
// Inter-chunk recurrence: S <- exp(lastA)*S + state; emit entry states (f16)
// and final_state (f32, to d_out tail).
// ---------------------------------------------------------------------------
__global__ void __launch_bounds__(256)
chunk_scan_kernel(const float* __restrict__ states, const float* __restrict__ init_state,
                  const float* __restrict__ acum, _Float16* __restrict__ Sin_h,
                  float* __restrict__ final_out) {
  int bh = blockIdx.x;  // b*32 + h
  for (int pos = threadIdx.x; pos < 8192; pos += 256) {
    float s = init_state[(size_t)bh * 8192 + pos];
    for (int ci = 0; ci < 32; ++ci) {
      Sin_h[((size_t)bh * 32 + ci) * 8192 + pos] = (_Float16)s;
      float lastA = acum[((size_t)bh * 32 + ci) * 64 + 63];
      s = __expf(lastA) * s + states[((size_t)bh * 32 + ci) * 8192 + pos];
    }
    final_out[(size_t)bh * 8192 + pos] = s;
  }
}

// ---------------------------------------------------------------------------
// Y_off[l,p] = exp(Ac[l]) * sum_n C[l,n]*S_in[p,n]; Y += Y_off + D*silu(x)
// ---------------------------------------------------------------------------
__global__ void __launch_bounds__(128)
ssd_yoff_kernel(const _Float16* __restrict__ Ch, const _Float16* __restrict__ Sin,
                const float* __restrict__ acum, const float* __restrict__ xconv,
                const float* __restrict__ Dp, float* __restrict__ Ybuf) {
  int bid = blockIdx.x;
  int h = bid & 31, ci = (bid >> 5) & 31, b = bid >> 10;
  __shared__ _Float16 Cld[64][136];   // [l][n]
  __shared__ _Float16 Sld[64][136];   // [p][n]
  __shared__ float Ac[64];
  int tid = threadIdx.x, lane = tid & 31, wid = tid >> 5;
  const float* ac = acum + (((size_t)(b * 32 + h)) * 32 + ci) * 64;
  if (tid < 64) Ac[tid] = ac[tid];
  const uint4* Cg4 = (const uint4*)(Ch + ((size_t)(b * 32 + ci)) * 8192);
  const uint4* Sg4 = (const uint4*)(Sin + (((size_t)(b * 32 + h)) * 32 + ci) * 8192);
  for (int idx = tid; idx < 1024; idx += 128) {
    int s = idx >> 3, n8 = (idx & 7) * 8;
    *(uint4*)&Cld[s][n8] = Cg4[idx];
    *(uint4*)&Sld[s][n8] = Sg4[idx];
  }
  __syncthreads();
  int L0 = wid * 16;
  v8f yo[4] = {};
  for (int n0 = 0; n0 < 128; n0 += 32) {
    v16h a = load_frag_a(&Cld[L0][n0], 136, lane);
#pragma unroll
    for (int t = 0; t < 4; ++t) {
      v16h bf = load_frag_b(&Sld[t * 16][n0], 136, lane);
      yo[t] = wmma_f32_16x16x32(a, bf, yo[t]);
    }
  }
  int pc = lane & 15, mb = (lane >> 4) * 8;
  float Dh = Dp[h];
#pragma unroll
  for (int t = 0; t < 4; ++t) {
    int p = t * 16 + pc;
#pragma unroll
    for (int r = 0; r < 8; ++r) {
      int l = L0 + mb + r;
      size_t row = (size_t)b * 2048 + ci * 64 + l;
      size_t yoff = row * 2048 + h * 64 + p;
      float xs = silu_f(xconv[row * 2304 + h * 64 + p]);
      Ybuf[yoff] = Ybuf[yoff] + __expf(Ac[l]) * yo[t][r] + Dh * xs;
    }
  }
}

// RMSNorm over d_inner, gate with sigmoid(z); f16 out for final GEMM
__global__ void __launch_bounds__(256)
rmsnorm_gate_kernel(const float* __restrict__ Y, const float* __restrict__ zx,
                    const float* __restrict__ norm_w, _Float16* __restrict__ out) {
  int row = blockIdx.x;  // b*L + l
  __shared__ float red[256];
  const float* y = Y + (size_t)row * 2048;
  float ss = 0.f;
  for (int d = threadIdx.x; d < 2048; d += 256) { float v = y[d]; ss += v * v; }
  red[threadIdx.x] = ss;
  __syncthreads();
  for (int off = 128; off > 0; off >>= 1) {
    if (threadIdx.x < off) red[threadIdx.x] += red[threadIdx.x + off];
    __syncthreads();
  }
  float rs = rsqrtf(red[0] * (1.f / 2048.f) + 1.1920929e-07f);
  const float* zr = zx + (size_t)row * 4384;
  for (int d = threadIdx.x; d < 2048; d += 256) {
    float v = y[d] * rs * norm_w[d];
    v *= 1.f / (1.f + __expf(-zr[d]));
    out[(size_t)row * 2048 + d] = (_Float16)v;
  }
}

// ---------------------------------------------------------------------------
extern "C" void kernel_launch(void* const* d_in, const int* in_sizes, int n_in,
                              void* d_out, int out_size, void* d_ws, size_t ws_size,
                              hipStream_t stream) {
  const float* x        = (const float*)d_in[0];  // (2,2048,1024)
  const float* state0   = (const float*)d_in[1];  // (2,32,64,128)
  const float* in_w     = (const float*)d_in[2];  // (4384,1024)
  const float* conv_w   = (const float*)d_in[3];  // (2304,4)
  const float* A_log    = (const float*)d_in[4];  // (32,)
  const float* dt_bias  = (const float*)d_in[5];  // (32,)
  const float* D_param  = (const float*)d_in[6];  // (32,)
  const float* norm_w   = (const float*)d_in[7];  // (2048,)
  const float* out_w    = (const float*)d_in[8];  // (1024,2048)
  float* out = (float*)d_out;
  float* final_state = out + (size_t)2 * 2048 * 1024;

  char* ws = (char*)d_ws;
  size_t o = 0;
  auto alloc = [&](size_t bytes) { size_t r = o; o += (bytes + 255) & ~(size_t)255; return r; };
  float*    zx     = (float*)   (ws + alloc((size_t)4096 * 4384 * 4));
  _Float16* xh     = (_Float16*)(ws + alloc((size_t)4096 * 1024 * 2));
  _Float16* wh     = (_Float16*)(ws + alloc((size_t)4384 * 1024 * 2));
  _Float16* oph    = (_Float16*)(ws + alloc((size_t)1024 * 2048 * 2));
  float*    xconv  = (float*)   (ws + alloc((size_t)4096 * 2304 * 4));
  _Float16* Xh     = (_Float16*)(ws + alloc((size_t)8388608 * 2));
  float*    dtbuf  = (float*)   (ws + alloc((size_t)4096 * 32 * 4));
  _Float16* Bh     = (_Float16*)(ws + alloc((size_t)524288 * 2));
  _Float16* Chh    = (_Float16*)(ws + alloc((size_t)524288 * 2));
  float*    acum   = (float*)   (ws + alloc((size_t)131072 * 4));
  float*    states = (float*)   (ws + alloc((size_t)16777216 * 4));
  _Float16* Sinh   = (_Float16*)(ws + alloc((size_t)16777216 * 2));
  float*    Ybuf   = (float*)   (ws + alloc((size_t)4096 * 2048 * 4));
  _Float16* ych    = (_Float16*)(ws + alloc((size_t)4096 * 2048 * 2));
  (void)ws_size; (void)n_in; (void)in_sizes; (void)out_size;

  auto blks = [](size_t n) { return (unsigned)((n + 255) / 256); };

  // 1. f32 -> f16 conversions
  f32_to_f16_kernel<<<blks((size_t)4096 * 1024), 256, 0, stream>>>(x, xh, (size_t)4096 * 1024);
  f32_to_f16_kernel<<<blks((size_t)4384 * 1024), 256, 0, stream>>>(in_w, wh, (size_t)4384 * 1024);
  f32_to_f16_kernel<<<blks((size_t)1024 * 2048), 256, 0, stream>>>(out_w, oph, (size_t)1024 * 2048);

  // 2. zxbcdt = x @ in_proj_w^T   (M=4096, N=4384, K=1024)
  gemm_f16_kernel<<<dim3(32, 35), 256, 0, stream>>>(xh, wh, zx, nullptr, 4096, 4384, 1024);

  // 3. conv + silu, X*dt, silu(B/C), cumsum(A*dt)
  conv_silu_kernel<<<blks((size_t)4096 * 2304), 256, 0, stream>>>(zx, conv_w, xconv);
  xdt_kernel<<<blks((size_t)4096 * 2048), 256, 0, stream>>>(zx, xconv, dt_bias, Xh, dtbuf);
  bc_silu_kernel<<<blks((size_t)4096 * 128), 256, 0, stream>>>(xconv, Bh, Chh);
  cumsum_kernel<<<2048, 64, 0, stream>>>(dtbuf, A_log, acum);

  // 4. intra-chunk SSD (WMMA)
  ssd_chunk_kernel<<<2048, 128, 0, stream>>>(Chh, Bh, Xh, acum, Ybuf, states);

  // 5. inter-chunk scan -> entry states (f16) + final_state (d_out tail)
  chunk_scan_kernel<<<64, 256, 0, stream>>>(states, state0, acum, Sinh, final_state);

  // 6. Y_off + D*silu(x) (WMMA)
  ssd_yoff_kernel<<<2048, 128, 0, stream>>>(Chh, Sinh, acum, xconv, D_param, Ybuf);

  // 7. RMSNorm + sigmoid(z) gate
  rmsnorm_gate_kernel<<<4096, 256, 0, stream>>>(Ybuf, zx, norm_w, ych);

  // 8. out = y_core @ out_proj_w^T + x   (M=4096, N=1024, K=2048)
  gemm_f16_kernel<<<dim3(32, 8), 256, 0, stream>>>(ych, oph, out, x, 4096, 1024, 2048);
}
